// Gemma3Main_15573551415418
// MI455X (gfx1250) — compile-verified, matrix operand order checked by reading, not run on hardware
//
#include <hip/hip_runtime.h>

// ---------------------------------------------------------------------------
// Gemma3-style 6-layer forward for MI455X (gfx1250, wave32, WMMA).
// Heavy math: v_wmma_f32_16x16x32_bf16, 64x64 per-wave tiles (16 WMMA/K-step,
// 1.0 ds_load_b128 per WMMA). bf16 operands staged via
// global_load_async_to_lds_b128 (ASYNCcnt); fp32 weights converted through
// VALU (v_cvt_pk_bf16_f32) into LDS.
// ---------------------------------------------------------------------------

typedef __bf16 bf16;
typedef __attribute__((ext_vector_type(16))) __bf16 v16bf;
typedef __attribute__((ext_vector_type(8)))  float  v8f;

constexpr int L_   = 6;
constexpr int T_   = 512;
constexpr int CTX_ = 1536;
constexpr int S_   = CTX_ + T_;   // 2048
constexpr int NH_  = 8;
constexpr int HD_  = 256;
constexpr int D_   = 2048;
constexpr int FF_  = 6144;
constexpr int V_   = 32000;
constexpr int QKV_ = 2560;        // (8+2)*256
constexpr float EPSf = 1e-6f;

__device__ __forceinline__ unsigned lds_addr_of(const void* p)
{
    // Generic LDS addresses carry the LDS byte offset in the low 32 bits
    // (ISA 10.2).
    return (unsigned)(uintptr_t)p;
}

__device__ __forceinline__ void async_ld_b128(unsigned lds, const void* gaddr)
{
    asm volatile("global_load_async_to_lds_b128 %0, %1, off"
                 :: "v"(lds), "v"(gaddr) : "memory");
}

__device__ __forceinline__ void wait_async0()
{
    asm volatile("s_wait_asynccnt 0x0" ::: "memory");
}

// ---------------------------------------------------------------------------
// Tiled WMMA GEMM: C[M,N] = A[M,K] (bf16, row-major) * B[N,K]^T
// B fp32 (weights, converted during LDS fill) or bf16 (K/V/attn, async DMA).
// Block: 256 threads = 8 waves. Tile: BM=128, BN=256, BK=32.
// Waves arranged 2 (row) x 4 (col); each wave owns 64x64 = acc[4][4].
// Callers guarantee M%128==0, N%256==0, K%32==0.
// ---------------------------------------------------------------------------
template <bool B_F32>
__global__ __launch_bounds__(256)
void gemm_wmma_kernel(const bf16* __restrict__ A, const void* __restrict__ Bv,
                      float* __restrict__ C, int N, int K,
                      long sAb, long sBb, long sCb,
                      int lda, int ldb, int ldc)
{
    constexpr int BM = 128, BN = 256, BK = 32;
    __shared__ bf16 sA[BM][BK];   // 8 KB
    __shared__ bf16 sB[BN][BK];   // 16 KB

    const int tid  = threadIdx.x;
    const int lane = tid & 31;
    const int wave = tid >> 5;
    const int wr   = wave & 1;            // 0..1 : 64-row group
    const int wc   = wave >> 1;           // 0..3 : 64-col group
    const int l15  = lane & 15;
    const int khalf = (lane >> 4) << 3;   // ISA 16-bit 16x32 A/B fragment layout
    const int rowh  = (lane >> 4) << 3;   // C layout: lanes>=16 hold M+8

    A += (long)blockIdx.z * sAb + (long)blockIdx.y * BM * lda;
    C += (long)blockIdx.z * sCb + (long)blockIdx.y * BM * ldc + (long)blockIdx.x * BN;
    const long bbase = (long)blockIdx.z * sBb + (long)blockIdx.x * BN * ldb;
    const float* Bf = (const float*)Bv;
    const bf16*  Bb = (const bf16*)Bv;

    // Per-thread tile-fill coordinates (16B chunks).
    const int fr = tid >> 2;              // 0..63 base row
    const int fc = (tid & 3) << 3;        // bf16 element col {0,8,16,24}
    const unsigned ldsA0 = lds_addr_of(&sA[fr][fc]);
    const unsigned ldsA1 = lds_addr_of(&sA[fr + 64][fc]);

    v8f acc[4][4] = {};

    for (int k0 = 0; k0 < K; k0 += BK) {
        // ---- A tile: 128x32 bf16, two async b128 chunks per thread ----
        async_ld_b128(ldsA0, A + (long)fr * lda + k0 + fc);
        async_ld_b128(ldsA1, A + (long)(fr + 64) * lda + k0 + fc);

        // ---- B tile: 256x32 ----
        if (B_F32) {
#pragma unroll
            for (int it = 0; it < 4; ++it) {
                int chunk = tid + it * 256;
                int r = chunk >> 2, c = (chunk & 3) << 3;
                const float* src = Bf + bbase + (long)r * ldb + k0 + c;
                float4 f0 = *(const float4*)src;
                float4 f1 = *(const float4*)(src + 4);
                bf16* dst = &sB[r][c];
                dst[0] = (bf16)f0.x; dst[1] = (bf16)f0.y;
                dst[2] = (bf16)f0.z; dst[3] = (bf16)f0.w;
                dst[4] = (bf16)f1.x; dst[5] = (bf16)f1.y;
                dst[6] = (bf16)f1.z; dst[7] = (bf16)f1.w;
            }
            if (k0 + BK < K)
                __builtin_prefetch(Bf + bbase + (long)fr * ldb + k0 + BK, 0, 1);
        } else {
#pragma unroll
            for (int it = 0; it < 4; ++it) {
                int chunk = tid + it * 256;
                int r = chunk >> 2, c = (chunk & 3) << 3;
                async_ld_b128(lds_addr_of(&sB[r][c]),
                              Bb + bbase + (long)r * ldb + k0 + c);
            }
        }
        wait_async0();
        __syncthreads();

        // ---- fragments: lane<16 holds K 0-7/16-23, lane>=16 holds 8-15/24-31 ----
        v16bf afr[4], bfr[4];
#pragma unroll
        for (int i = 0; i < 4; ++i) {
            const bf16* p = &sA[wr * 64 + i * 16 + l15][khalf];
            v16bf t;
#pragma unroll
            for (int e = 0; e < 8; ++e) { t[e] = p[e]; t[8 + e] = p[16 + e]; }
            afr[i] = t;
        }
#pragma unroll
        for (int j = 0; j < 4; ++j) {
            const bf16* p = &sB[wc * 64 + j * 16 + l15][khalf];
            v16bf t;
#pragma unroll
            for (int e = 0; e < 8; ++e) { t[e] = p[e]; t[8 + e] = p[16 + e]; }
            bfr[j] = t;
        }
#pragma unroll
        for (int i = 0; i < 4; ++i)
#pragma unroll
            for (int j = 0; j < 4; ++j)
                acc[i][j] = __builtin_amdgcn_wmma_f32_16x16x32_bf16(
                    false, afr[i], false, bfr[j], (short)0, acc[i][j], false, false);
        __syncthreads();
    }

    // ---- store C per 16x16 f32 D layout (VGPR r: M=r / M=r+8) ----
#pragma unroll
    for (int i = 0; i < 4; ++i)
#pragma unroll
        for (int j = 0; j < 4; ++j) {
            int col = wc * 64 + j * 16 + l15;
#pragma unroll
            for (int r = 0; r < 8; ++r) {
                int row = wr * 64 + i * 16 + rowh + r;
                C[(long)row * ldc + col] = acc[i][j][r];
            }
        }
}

// ---------------------------------------------------------------------------
// RMSNorm row -> bf16 : out = x * rsqrt(mean(x^2)+eps) * (1+w)
// ---------------------------------------------------------------------------
__global__ __launch_bounds__(256)
void rmsnorm_bf16_kernel(const float* __restrict__ x, const float* __restrict__ w,
                         bf16* __restrict__ out, int dim)
{
    const int t = blockIdx.x;
    x   += (long)t * dim;
    out += (long)t * dim;
    __shared__ float red[256];
    float s = 0.f;
    for (int i = threadIdx.x; i < dim; i += 256) { float v = x[i]; s += v * v; }
    red[threadIdx.x] = s; __syncthreads();
    for (int o = 128; o > 0; o >>= 1) {
        if (threadIdx.x < o) red[threadIdx.x] += red[threadIdx.x + o];
        __syncthreads();
    }
    float r = rsqrtf(red[0] / (float)dim + EPSf);
    for (int i = threadIdx.x; i < dim; i += 256)
        out[i] = (bf16)(x[i] * r * (1.f + w[i]));
}

// h += rms(x, w)
__global__ __launch_bounds__(256)
void residual_rms_add_kernel(float* __restrict__ h, const float* __restrict__ x,
                             const float* __restrict__ w, int dim)
{
    const int t = blockIdx.x;
    h += (long)t * dim;
    x += (long)t * dim;
    __shared__ float red[256];
    float s = 0.f;
    for (int i = threadIdx.x; i < dim; i += 256) { float v = x[i]; s += v * v; }
    red[threadIdx.x] = s; __syncthreads();
    for (int o = 128; o > 0; o >>= 1) {
        if (threadIdx.x < o) red[threadIdx.x] += red[threadIdx.x + o];
        __syncthreads();
    }
    float r = rsqrtf(red[0] / (float)dim + EPSf);
    for (int i = threadIdx.x; i < dim; i += 256)
        h[i] += x[i] * r * (1.f + w[i]);
}

// ---------------------------------------------------------------------------
// Per-(t, head) RMSNorm + RoPE for q (blockIdx.y < NH) and k (blockIdx.y == NH).
// ---------------------------------------------------------------------------
__global__ __launch_bounds__(HD_)
void qk_norm_rope_kernel(const float* __restrict__ qkv, const float* __restrict__ qw,
                         const float* __restrict__ kw, const float* __restrict__ cosx,
                         const float* __restrict__ sinx, bf16* __restrict__ qout,
                         bf16* __restrict__ kfull, float* __restrict__ kslice)
{
    const int t = blockIdx.x, hh = blockIdx.y, d = threadIdx.x;
    const float* row = qkv + (long)t * QKV_ + (hh < NH_ ? hh * HD_ : NH_ * HD_);
    const float* w = (hh < NH_) ? qw : kw;
    __shared__ float red[HD_];
    __shared__ float xs[HD_];
    float v = row[d];
    red[d] = v * v; __syncthreads();
    for (int o = HD_ / 2; o > 0; o >>= 1) {
        if (d < o) red[d] += red[d + o];
        __syncthreads();
    }
    float r   = rsqrtf(red[0] / (float)HD_ + EPSf);
    float xnv = v * r * (1.f + w[d]);
    xs[d] = xnv; __syncthreads();
    float xr = (d < HD_ / 2) ? -xs[d + HD_ / 2] : xs[d - HD_ / 2];
    float ov = xnv * cosx[(long)t * HD_ + d] + xr * sinx[(long)t * HD_ + d];
    if (hh < NH_) {
        qout[((long)hh * T_ + t) * HD_ + d] = (bf16)ov;
    } else {
        kfull[(long)(CTX_ + t) * HD_ + d] = (bf16)ov;
        kslice[(long)t * HD_ + d] = ov;
    }
}

// v row t -> vfull[:, CTX+t] (bf16, [HD,S]) and fp32 v_slice [HD,T]
__global__ __launch_bounds__(HD_)
void v_prep_kernel(const float* __restrict__ qkv, bf16* __restrict__ vfull,
                   float* __restrict__ vslice)
{
    const int t = blockIdx.x, d = threadIdx.x;
    float v = qkv[(long)t * QKV_ + (NH_ + 1) * HD_ + d];
    vfull[(long)d * S_ + CTX_ + t] = (bf16)v;
    vslice[(long)d * T_ + t] = v;
}

// fp32 KV cache -> bf16 working buffers
__global__ void kcache_kernel(const float* __restrict__ kv_k, bf16* __restrict__ kfull)
{
    long i = (long)blockIdx.x * 256 + threadIdx.x;
    if (i < (long)CTX_ * HD_) kfull[i] = (bf16)kv_k[i];   // rows 0..CTX-1 of [S,HD]
}
__global__ void vcache_kernel(const float* __restrict__ kv_v, bf16* __restrict__ vfull)
{
    long i = (long)blockIdx.x * 256 + threadIdx.x;
    if (i < (long)HD_ * CTX_) {
        int d = (int)(i / CTX_), c = (int)(i % CTX_);
        vfull[(long)d * S_ + c] = (bf16)kv_v[i];
    }
}

// scores -> softcap + mask + softmax -> bf16 attention probabilities
__global__ __launch_bounds__(256)
void softmax_kernel(float* __restrict__ scores, const float* __restrict__ mask,
                    bf16* __restrict__ attn, float qk_scale)
{
    const int t = blockIdx.x, hh = blockIdx.y, tid = threadIdx.x;
    float* row = scores + ((long)hh * T_ + t) * S_;
    const float* mrow = mask + (long)t * S_;
    bf16* arow = attn + ((long)hh * T_ + t) * S_;
    __shared__ float red[256];
    float m = -1e30f;
    for (int i = tid; i < S_; i += 256) {
        float x = row[i] * qk_scale;
        x = tanhf(x * (1.f / 50.f)) * 50.f + mrow[i];
        row[i] = x;
        m = fmaxf(m, x);
    }
    red[tid] = m; __syncthreads();
    for (int o = 128; o > 0; o >>= 1) {
        if (tid < o) red[tid] = fmaxf(red[tid], red[tid + o]);
        __syncthreads();
    }
    m = red[0]; __syncthreads();
    float s = 0.f;
    for (int i = tid; i < S_; i += 256) {
        float e = __expf(row[i] - m);
        row[i] = e;
        s += e;
    }
    red[tid] = s; __syncthreads();
    for (int o = 128; o > 0; o >>= 1) {
        if (tid < o) red[tid] += red[tid + o];
        __syncthreads();
    }
    float inv = 1.f / red[0];
    for (int i = tid; i < S_; i += 256) arow[i] = (bf16)(row[i] * inv);
}

__global__ void f32_to_bf16_kernel(const float* __restrict__ in, bf16* __restrict__ out, long n)
{
    for (long i = (long)blockIdx.x * 256 + threadIdx.x; i < n; i += (long)gridDim.x * 256)
        out[i] = (bf16)in[i];
}

__device__ __forceinline__ float gelu_tanh(float x)
{
    return 0.5f * x * (1.f + tanhf(0.7978845608f * (x + 0.044715f * x * x * x)));
}

__global__ void gelu_mul_kernel(const float* __restrict__ g, const float* __restrict__ u,
                                bf16* __restrict__ out, long n)
{
    long i = (long)blockIdx.x * 256 + threadIdx.x;
    if (i < n) out[i] = (bf16)(gelu_tanh(g[i]) * u[i]);
}

// ---------------------------------------------------------------------------
// Host orchestration
// ---------------------------------------------------------------------------
extern "C" void kernel_launch(void* const* d_in, const int* in_sizes, int n_in,
                              void* d_out, int out_size, void* d_ws, size_t ws_size,
                              hipStream_t stream)
{
    const float* embeddings   = (const float*)d_in[0];
    const float* mask_global  = (const float*)d_in[1];
    const float* mask_local   = (const float*)d_in[2];
    const float* pos_cos      = (const float*)d_in[3];
    const float* pos_sin      = (const float*)d_in[4];
    const float* pos_lcos     = (const float*)d_in[5];
    const float* pos_lsin     = (const float*)d_in[6];
    const float* kv_k         = (const float*)d_in[7];
    const float* kv_v         = (const float*)d_in[8];
    const float* pre_attn_w   = (const float*)d_in[9];
    const float* qkv_w        = (const float*)d_in[10];
    const float* q_norm_w     = (const float*)d_in[11];
    const float* k_norm_w     = (const float*)d_in[12];
    const float* out_w        = (const float*)d_in[13];
    const float* post_attn_w  = (const float*)d_in[14];
    const float* pre_ff_w     = (const float*)d_in[15];
    const float* w1_w         = (const float*)d_in[16];
    const float* w3_w         = (const float*)d_in[17];
    const float* w2_w         = (const float*)d_in[18];
    const float* post_ff_w    = (const float*)d_in[19];
    const float* final_norm_w = (const float*)d_in[20];
    const float* lm_head_w    = (const float*)d_in[21];
    (void)in_sizes; (void)n_in; (void)out_size; (void)ws_size;

    // ---- workspace carve (aliased regions: scores|g/up, attn|gelu*up) ----
    char* ws = (char*)d_ws;
    size_t off = 0;
    auto carve = [&](size_t bytes) -> void* {
        void* p = ws + off;
        off += (bytes + 255) & ~(size_t)255;
        return p;
    };
    float* h     = (float*)carve((size_t)T_ * D_ * 4);
    bf16*  xn    = (bf16*) carve((size_t)T_ * D_ * 2);
    float* qkvb  = (float*)carve((size_t)T_ * QKV_ * 4);
    bf16*  qb    = (bf16*) carve((size_t)NH_ * T_ * HD_ * 2);
    bf16*  kfull = (bf16*) carve((size_t)S_ * HD_ * 2);
    bf16*  vfull = (bf16*) carve((size_t)HD_ * S_ * 2);
    char*  regA  = (char*) carve((size_t)NH_ * T_ * S_ * 4);  // 33.5 MB
    float* scores = (float*)regA;                              // attention phase
    float* gbuf   = (float*)regA;                              // FFN phase
    float* ubuf   = (float*)(regA + (size_t)T_ * FF_ * 4);
    char*  regB  = (char*) carve((size_t)NH_ * T_ * S_ * 2);  // 16.8 MB
    bf16*  attnb = (bf16*)regB;
    bf16*  gu    = (bf16*)regB;
    float* ao    = (float*)carve((size_t)T_ * NH_ * HD_ * 4);
    bf16*  aob   = (bf16*) carve((size_t)T_ * NH_ * HD_ * 2);
    float* proj  = (float*)carve((size_t)T_ * D_ * 4);

    float* logits      = (float*)d_out;
    float* kslice_base = logits + (size_t)T_ * V_;
    float* vslice_base = kslice_base + (size_t)L_ * T_ * HD_;

    hipMemcpyAsync(h, embeddings, (size_t)T_ * D_ * 4, hipMemcpyDeviceToDevice, stream);

    for (int i = 0; i < L_; ++i) {
        const bool is_local = ((i + 1) % 6) == 0;
        const float* cosx = is_local ? pos_lcos : pos_cos;
        const float* sinx = is_local ? pos_lsin : pos_sin;
        const float* mask = is_local ? mask_local : mask_global;

        // ---- attention block ----
        rmsnorm_bf16_kernel<<<T_, 256, 0, stream>>>(h, pre_attn_w + (size_t)i * D_, xn, D_);
        gemm_wmma_kernel<true><<<dim3(QKV_ / 256, T_ / 128, 1), 256, 0, stream>>>(
            xn, qkv_w + (size_t)i * QKV_ * D_, qkvb, QKV_, D_, 0, 0, 0, D_, D_, QKV_);
        kcache_kernel<<<(CTX_ * HD_ + 255) / 256, 256, 0, stream>>>(
            kv_k + (size_t)i * CTX_ * HD_, kfull);
        vcache_kernel<<<(HD_ * CTX_ + 255) / 256, 256, 0, stream>>>(
            kv_v + (size_t)i * HD_ * CTX_, vfull);
        qk_norm_rope_kernel<<<dim3(T_, NH_ + 1), HD_, 0, stream>>>(
            qkvb, q_norm_w + (size_t)i * HD_, k_norm_w + (size_t)i * HD_,
            cosx, sinx, qb, kfull, kslice_base + (size_t)i * T_ * HD_);
        v_prep_kernel<<<T_, HD_, 0, stream>>>(qkvb, vfull, vslice_base + (size_t)i * HD_ * T_);

        // scores[h] = Q[h] (T,HD) x Kfull (S,HD)^T  — batched over heads
        gemm_wmma_kernel<false><<<dim3(S_ / 256, T_ / 128, NH_), 256, 0, stream>>>(
            qb, kfull, scores, S_, HD_, (long)T_ * HD_, 0, (long)T_ * S_, HD_, HD_, S_);
        softmax_kernel<<<dim3(T_, NH_), 256, 0, stream>>>(scores, mask, attnb, 0.0625f);
        // ao[h] = attn[h] (T,S) x Vfull (HD,S)^T, written into [T, NH*HD] at col h*HD
        gemm_wmma_kernel<false><<<dim3(HD_ / 256, T_ / 128, NH_), 256, 0, stream>>>(
            attnb, vfull, ao, HD_, S_, (long)T_ * S_, 0, (long)HD_, S_, S_, NH_ * HD_);
        f32_to_bf16_kernel<<<1024, 256, 0, stream>>>(ao, aob, (long)T_ * NH_ * HD_);
        gemm_wmma_kernel<true><<<dim3(D_ / 256, T_ / 128, 1), 256, 0, stream>>>(
            aob, out_w + (size_t)i * D_ * (NH_ * HD_), proj, D_, NH_ * HD_,
            0, 0, 0, NH_ * HD_, NH_ * HD_, D_);
        residual_rms_add_kernel<<<T_, 256, 0, stream>>>(h, proj, post_attn_w + (size_t)i * D_, D_);

        // ---- FFN block ----
        rmsnorm_bf16_kernel<<<T_, 256, 0, stream>>>(h, pre_ff_w + (size_t)i * D_, xn, D_);
        gemm_wmma_kernel<true><<<dim3(FF_ / 256, T_ / 128, 1), 256, 0, stream>>>(
            xn, w1_w + (size_t)i * FF_ * D_, gbuf, FF_, D_, 0, 0, 0, D_, D_, FF_);
        gemm_wmma_kernel<true><<<dim3(FF_ / 256, T_ / 128, 1), 256, 0, stream>>>(
            xn, w3_w + (size_t)i * FF_ * D_, ubuf, FF_, D_, 0, 0, 0, D_, D_, FF_);
        gelu_mul_kernel<<<((long)T_ * FF_ + 255) / 256, 256, 0, stream>>>(
            gbuf, ubuf, gu, (long)T_ * FF_);
        gemm_wmma_kernel<true><<<dim3(D_ / 256, T_ / 128, 1), 256, 0, stream>>>(
            gu, w2_w + (size_t)i * D_ * FF_, proj, D_, FF_, 0, 0, 0, FF_, FF_, D_);
        residual_rms_add_kernel<<<T_, 256, 0, stream>>>(h, proj, post_ff_w + (size_t)i * D_, D_);
    }

    // ---- final norm + lm_head ----
    rmsnorm_bf16_kernel<<<T_, 256, 0, stream>>>(h, final_norm_w, xn, D_);
    gemm_wmma_kernel<true><<<dim3(V_ / 256, T_ / 128, 1), 256, 0, stream>>>(
        xn, lm_head_w, logits, V_, D_, 0, 0, 0, D_, D_, V_);
}
